// SkipConnection_52166672777658
// MI455X (gfx1250) — matrix operand order repaired and checked by reading, not compiled, not run on hardware
//
#include <hip/hip_runtime.h>
#include <hip/hip_bf16.h>

// ---------------------------------------------------------------------------
// DeepViT-style re-attention block for MI455X (gfx1250, wave32, WMMA).
// B=8, N=1024, DIM=768, HEADS=8, HEAD_DIM=96, S=16, CH=3.
// All GEMMs use v_wmma_f32_16x16x32_bf16; operands are stored so every
// fragment is two 128-bit loads (A: row-major; B: column-major == K-contig).
// Conv image staging uses GLOBAL_LOAD_ASYNC_TO_LDS_B128 (ASYNCcnt domain).
// ---------------------------------------------------------------------------

#define Bsz      8
#define Nseq     1024
#define DIM      768
#define HEADS    8
#define HDIM     96
#define PST      1040   // padded LDS row stride (elements) to dodge bank conflicts

typedef __attribute__((ext_vector_type(16))) __bf16        bf16x16;
typedef __attribute__((ext_vector_type(8)))  float         f32x8;
typedef __attribute__((ext_vector_type(4)))  unsigned int  u32x4;

union Frag16 { bf16x16 v; u32x4 q[2]; };

__device__ __forceinline__ f32x8 wmma_bf16(bf16x16 a, bf16x16 b, f32x8 c) {
  return __builtin_amdgcn_wmma_f32_16x16x32_bf16(false, a, false, b, (short)0, c,
                                                 false, false);
}

// round-to-nearest-even float -> bf16 bits (pure integer math)
__device__ __forceinline__ unsigned short f2bf(float f) {
  unsigned int u = __builtin_bit_cast(unsigned int, f);
  u += 0x7FFFu + ((u >> 16) & 1u);
  return (unsigned short)(u >> 16);
}
__device__ __forceinline__ float bf2f(unsigned short s) {
  unsigned int u = ((unsigned int)s) << 16;
  return __builtin_bit_cast(float, u);
}

// A fragment (16x32, 16-bit): lane r = row; K = 2*kb8+0..15 contiguous.
__device__ __forceinline__ Frag16 load_a(const unsigned short* row, int kb8) {
  Frag16 f;
  f.q[0] = *(const u32x4*)(row + 2 * kb8);
  f.q[1] = *(const u32x4*)(row + 2 * kb8 + 8);
  return f;
}
// B fragment (32x16, 16-bit): lane = column; K = kb16+0..15 contiguous.
__device__ __forceinline__ Frag16 load_b(const unsigned short* col) {
  Frag16 f;
  f.q[0] = *(const u32x4*)(col);
  f.q[1] = *(const u32x4*)(col + 8);
  return f;
}

// Async global->LDS 16-byte copy (ASYNCcnt domain, no VGPR round-trip).
__device__ __forceinline__ void async_g2l_b128(unsigned ldsByteAddr,
                                               const void* gptr) {
  unsigned long long ga = (unsigned long long)(size_t)gptr;
  asm volatile("global_load_async_to_lds_b128 %0, %1, off"
               :: "v"(ldsByteAddr), "v"(ga)
               : "memory");
}
__device__ __forceinline__ void wait_asynccnt0() {
  asm volatile("s_wait_asynccnt 0x0" ::: "memory");
}

// ---------------------------------------------------------------------------
// Kernel 1: 3x3 SAME conv on 16x16x3 token images, split heads, emit bf16:
//   Qh: (B,H,N,96)  row-major  (A operand of Q@K^T)
//   Kh: (B,H,N,96)  row-major  (B operand of Q@K^T: col-major of d x m)
//   Vt: (B,H,96,N)  row-major  (B operand of P@V:  col-major of m x d)
// ---------------------------------------------------------------------------
__global__ __launch_bounds__(256) void conv_split_kernel(
    const float* __restrict__ q, const float* __restrict__ k,
    const float* __restrict__ v, const float* __restrict__ conv_w,
    unsigned short* __restrict__ Qh, unsigned short* __restrict__ Kh,
    unsigned short* __restrict__ Vt) {
  __shared__ __align__(16) float img[3][DIM];    // 9216 B, staged via async DMA
  __shared__ float w[81];
  const int tok = blockIdx.x;            // b*N + n
  const int b = tok >> 10, n = tok & 1023;
  const float* src[3] = {q, k, v};
  const int t = threadIdx.x;
  if (t < 81) w[t] = conv_w[t];
  // 576 x 16-byte chunks: global -> LDS, async (ASYNCcnt), then wait.
  {
    const unsigned ldsBase = (unsigned)(size_t)(&img[0][0]);
    for (int i = t; i < 576; i += 256) {
      int ts = i / 192, c16 = i % 192;                  // 192 chunks per tensor
      async_g2l_b128(ldsBase + (unsigned)i * 16u,
                     src[ts] + (size_t)tok * DIM + c16 * 4);
    }
    wait_asynccnt0();
  }
  __syncthreads();
  for (int i = t; i < 3 * DIM; i += 256) {
    int ts = i / DIM, p = i % DIM;
    int co = p % 3, xy = p / 3, x = xy & 15, y = xy >> 4;
    float acc = 0.f;
#pragma unroll
    for (int ky = 0; ky < 3; ky++) {
      int yy = y + ky - 1;
      if ((unsigned)yy >= 16u) continue;
#pragma unroll
      for (int kx = 0; kx < 3; kx++) {
        int xx = x + kx - 1;
        if ((unsigned)xx >= 16u) continue;
        const float* ip = &img[ts][(yy * 16 + xx) * 3];
        const float* wp = &w[(ky * 3 + kx) * 9 + co];     // [ky][kx][ci][co]
        acc += ip[0] * wp[0] + ip[1] * wp[3] + ip[2] * wp[6];
      }
    }
    int h = p / HDIM, d = p % HDIM;
    unsigned short bv = f2bf(acc);
    size_t bh = (size_t)(b * HEADS + h);
    if (ts == 0)      Qh[(bh * Nseq + n) * HDIM + d] = bv;
    else if (ts == 1) Kh[(bh * Nseq + n) * HDIM + d] = bv;
    else              Vt[(bh * HDIM + d) * Nseq + n] = bv;
  }
}

// ---------------------------------------------------------------------------
// Kernel 1b: fp32 -> bf16 convert + transpose of proj_w:
//   proj_w (k,n) row-major  ->  Wt (n,k) row-major (K-contiguous B operand)
// ---------------------------------------------------------------------------
__global__ __launch_bounds__(256) void cvt_w_kernel(const float* __restrict__ w,
                                                    unsigned short* __restrict__ wt) {
  int i = blockIdx.x * blockDim.x + threadIdx.x;
  if (i < DIM * DIM) {
    int kk = i / DIM, nn = i % DIM;
    wt[nn * DIM + kk] = f2bf(w[i]);
  }
}

// ---------------------------------------------------------------------------
// Kernel 2: fused attention + re-attention (LDS-resident probabilities).
// One block = one (b, 16-row query tile). 16 waves (512 threads).
// LDS: P[8 heads][16 rows][PST] bf16 (~260 KB of the 320 KB WGP LDS).
// phase1: wave w -> head w/2, m-half w%2: scores via WMMA, bf16 -> LDS
// phase2: row softmax (128 rows, one per thread)
// phase3: in-place head mixing with BN folded: P_g = sum_h c[h][g]*A_h + e[g]
//         (the constant e[g] flows through P@V as e[g]*colsum(V) exactly)
// phase4: wave w -> out-head w/2, d-half w%2: O = P_g @ V_g via WMMA
// ---------------------------------------------------------------------------
__global__ __launch_bounds__(512) void attn_kernel(
    const unsigned short* __restrict__ Qh, const unsigned short* __restrict__ Kh,
    const unsigned short* __restrict__ Vt,
    const float* __restrict__ reatten_w, const float* __restrict__ reatten_b,
    const float* __restrict__ bn_gamma, const float* __restrict__ bn_beta,
    const float* __restrict__ bn_mean, const float* __restrict__ bn_var,
    unsigned short* __restrict__ Obf) {
  extern __shared__ unsigned short P[];   // [HEADS][16][PST]
  __shared__ float cC[HEADS][HEADS];      // c[h][g] = w[h,g] * gamma_g/sqrt(var_g+eps)
  __shared__ float eC[HEADS];             // e[g]    = (b_g - mean_g)*inv_g + beta_g

  const int tid = threadIdx.x, lane = tid & 31, wave = tid >> 5;
  const int blk = blockIdx.x;             // b*64 + ntile
  const int b = blk >> 6, n0 = (blk & 63) << 4;

  if (tid < 64) {
    int h = tid >> 3, g = tid & 7;
    float inv = bn_gamma[g] * rsqrtf(bn_var[g] + 1e-3f);
    cC[h][g] = reatten_w[h * 8 + g] * inv;
    if (h == 0) eC[g] = (reatten_b[g] - bn_mean[g]) * inv + bn_beta[g];
  }

  const int r     = lane & 15;            // A-row / B,C-column within tile
  const int kb8   = (lane < 16) ? 0 : 8;  // A-operand K-base
  const int kb16  = (lane < 16) ? 0 : 16; // B-operand K-base
  const int rbase = (lane < 16) ? 0 : 8;  // C/D row base

  // ---- phase 1: S_h = (Q_h K_h^T) * scale -> bf16 in LDS --------------------
  {
    const int h = wave >> 1, mhalf = wave & 1;
    const float scale = 0.1020620726159657f;   // 96^-0.5
    const size_t bh = (size_t)(b * HEADS + h);
    const unsigned short* Qrow = Qh + (bh * Nseq + n0 + r) * HDIM;
    const unsigned short* Kb   = Kh + bh * Nseq * HDIM;
    Frag16 aq[3];
#pragma unroll
    for (int c = 0; c < 3; c++) aq[c] = load_a(Qrow + c * 32, kb8);

    unsigned short* pr = P + (size_t)(h * 16) * PST;
    for (int t = 0; t < 32; t++) {
      const int m0 = mhalf * 512 + t * 16;
      const unsigned short* Krow = Kb + (size_t)(m0 + r) * HDIM;  // col m per lane
      __builtin_prefetch(Krow + 16 * HDIM, 0, 1);    // global_prefetch_b8
      f32x8 acc = {};
#pragma unroll
      for (int c = 0; c < 3; c++) {
        Frag16 bk = load_b(Krow + c * 32 + kb16);
        acc = wmma_bf16(aq[c].v, bk.v, acc);
      }
#pragma unroll
      for (int vv = 0; vv < 8; vv++)
        pr[(size_t)(rbase + vv) * PST + (m0 + r)] = f2bf(acc[vv] * scale);
    }
  }
  __syncthreads();

  // ---- phase 2: softmax over each of the 128 (head,row) rows ----------------
  if (tid < 128) {
    unsigned short* row = P + (size_t)tid * PST;
    float mx = -3.0e38f;
#pragma unroll 1
    for (int m = 0; m < Nseq; m++) mx = fmaxf(mx, bf2f(row[m]));
    float sum = 0.f;
#pragma unroll 1
    for (int m = 0; m < Nseq; m++) {
      float e = __expf(bf2f(row[m]) - mx);
      sum += e;
      row[m] = f2bf(e);
    }
    float rinv = 1.0f / sum;
#pragma unroll 1
    for (int m = 0; m < Nseq; m++) row[m] = f2bf(bf2f(row[m]) * rinv);
  }
  __syncthreads();

  // ---- phase 3: in-place re-attention head mix (+BN fold) -------------------
#pragma unroll 1
  for (int j = 0; j < 32; j++) {
    int idx = tid + j * 512;                // 16 rows x 1024 cols
    int nr = idx >> 10, m = idx & 1023;
    float a[HEADS];
#pragma unroll
    for (int hh = 0; hh < HEADS; hh++)
      a[hh] = bf2f(P[(size_t)(hh * 16 + nr) * PST + m]);
#pragma unroll
    for (int g = 0; g < HEADS; g++) {
      float p = eC[g];
#pragma unroll
      for (int hh = 0; hh < HEADS; hh++) p += a[hh] * cC[hh][g];
      P[(size_t)(g * 16 + nr) * PST + m] = f2bf(p);
    }
  }
  __syncthreads();

  // ---- phase 4: O_g = P_g @ V_g --------------------------------------------
  {
    const int g = wave >> 1, dhalf = wave & 1;
    const size_t bg = (size_t)(b * HEADS + g);
    const unsigned short* Prow = P + (size_t)(g * 16 + r) * PST;
    f32x8 oacc[3] = {{}, {}, {}};
    for (int kc = 0; kc < 32; kc++) {
      const int m0 = kc * 32;
      Frag16 ap = load_a(Prow + m0, kb8);           // ds_load_b128 x2
#pragma unroll
      for (int dt = 0; dt < 3; dt++) {
        const int d0 = dhalf * 48 + dt * 16;
        const unsigned short* vp =
            Vt + ((size_t)bg * HDIM + d0 + r) * Nseq + m0 + kb16;
        __builtin_prefetch(vp + 32, 0, 1);          // global_prefetch_b8
        Frag16 bv = load_b(vp);                     // global_load_b128 x2
        oacc[dt] = wmma_bf16(ap.v, bv.v, oacc[dt]);
      }
    }
    unsigned short* Ob = Obf + ((size_t)(b * Nseq + n0)) * DIM + g * HDIM;
#pragma unroll
    for (int dt = 0; dt < 3; dt++) {
      const int d0 = dhalf * 48 + dt * 16 + r;
#pragma unroll
      for (int vv = 0; vv < 8; vv++)
        Ob[(size_t)(rbase + vv) * DIM + d0] = f2bf(oacc[dt][vv]);
    }
  }
}

// ---------------------------------------------------------------------------
// Kernel 3: OUT(8192x768) = O_bf16 @ W + bias (fp32 out), W as Wt(n,k) bf16.
// Block = 8 waves; each wave owns one 16x16 output tile, K=768 (24 WMMAs).
// ---------------------------------------------------------------------------
__global__ __launch_bounds__(256) void proj_kernel(
    const unsigned short* __restrict__ Obf, const unsigned short* __restrict__ Wt,
    const float* __restrict__ proj_b, float* __restrict__ out) {
  const int lane = threadIdx.x & 31, wave = threadIdx.x >> 5;
  const int mt = blockIdx.x;              // 0..511 (16-row tiles of B*N)
  const int nt = blockIdx.y * 8 + wave;   // 0..47  (16-col tiles of DIM)
  const int r = lane & 15;
  const int kb8 = (lane < 16) ? 0 : 8, kb16 = (lane < 16) ? 0 : 16;
  const int rbase = (lane < 16) ? 0 : 8;

  const unsigned short* Arow = Obf + ((size_t)mt * 16 + r) * DIM;
  const unsigned short* Brow = Wt + (size_t)(nt * 16 + r) * DIM;  // col n per lane
  f32x8 acc = {};
  for (int kc = 0; kc < 24; kc++) {
    const int k0 = kc * 32;
    __builtin_prefetch(Brow + k0 + 64, 0, 1);
    Frag16 a  = load_a(Arow + k0, kb8);
    Frag16 bw = load_b(Brow + k0 + kb16);
    acc = wmma_bf16(a.v, bw.v, acc);
  }
  const float bias = proj_b[nt * 16 + r];
  float* op = out + ((size_t)mt * 16) * DIM + nt * 16 + r;
#pragma unroll
  for (int vv = 0; vv < 8; vv++)
    op[(size_t)(rbase + vv) * DIM] = acc[vv] + bias;
}

// ---------------------------------------------------------------------------
extern "C" void kernel_launch(void* const* d_in, const int* in_sizes, int n_in,
                              void* d_out, int out_size, void* d_ws, size_t ws_size,
                              hipStream_t stream) {
  (void)in_sizes; (void)n_in; (void)out_size; (void)ws_size;
  const float* q         = (const float*)d_in[0];
  const float* k         = (const float*)d_in[1];
  const float* v         = (const float*)d_in[2];
  const float* conv_w    = (const float*)d_in[3];
  const float* reatten_w = (const float*)d_in[4];
  const float* reatten_b = (const float*)d_in[5];
  const float* bn_gamma  = (const float*)d_in[6];
  const float* bn_beta   = (const float*)d_in[7];
  const float* bn_mean   = (const float*)d_in[8];
  const float* bn_var    = (const float*)d_in[9];
  const float* proj_w    = (const float*)d_in[10];
  const float* proj_b    = (const float*)d_in[11];

  char* ws = (char*)d_ws;
  const size_t szHeads = (size_t)Bsz * HEADS * Nseq * HDIM * 2;   // 12.58 MB bf16
  unsigned short* Qh  = (unsigned short*)(ws);
  unsigned short* Kh  = (unsigned short*)(ws + szHeads);
  unsigned short* Vt  = (unsigned short*)(ws + 2 * szHeads);
  unsigned short* Obf = (unsigned short*)(ws + 3 * szHeads);
  unsigned short* Wt  = (unsigned short*)(ws + 4 * szHeads);      // 768*768 bf16

  conv_split_kernel<<<Bsz * Nseq, 256, 0, stream>>>(q, k, v, conv_w, Qh, Kh, Vt);
  cvt_w_kernel<<<(DIM * DIM + 255) / 256, 256, 0, stream>>>(proj_w, Wt);

  const size_t ldsBytes = (size_t)HEADS * 16 * PST * 2;           // ~260 KB / WGP
  hipFuncSetAttribute((const void*)attn_kernel,
                      hipFuncAttributeMaxDynamicSharedMemorySize, (int)ldsBytes);
  attn_kernel<<<Bsz * (Nseq / 16), 512, ldsBytes, stream>>>(
      Qh, Kh, Vt, reatten_w, reatten_b, bn_gamma, bn_beta, bn_mean, bn_var, Obf);

  proj_kernel<<<dim3(Bsz * Nseq / 16, 6), 256, 0, stream>>>(Obf, Wt, proj_b,
                                                            (float*)d_out);
}
